// MoleculeNetRegressor_66125316489287
// MI455X (gfx1250) — matrix operand order repaired and checked by reading, not compiled, not run on hardware
//
#include <hip/hip_runtime.h>
#include <hip/hip_bf16.h>

#define NODES  100000
#define EDGES  1600000
#define FIN    128
#define HID    64
#define NGRAPH 2048
#define BN_EPS 1e-5f
#define ROWT   (NODES / 16)   // 6250 row tiles

typedef __attribute__((ext_vector_type(16))) __bf16 v16bf;
typedef __attribute__((ext_vector_type(8)))  float  v8f;

// ---------------------------------------------------------------- utilities
__global__ void fill_f32(float* p, float v, int n) {
    int i = blockIdx.x * blockDim.x + threadIdx.x;
    if (i < n) p[i] = v;
}
__global__ void fill_i32(int* p, int v, int n) {
    int i = blockIdx.x * blockDim.x + threadIdx.x;
    if (i < n) p[i] = v;
}

// Pre-swizzle W[K x 64] (fp32) into per-(strip, chunk, lane) contiguous bf16
// fragments: Bs[((w*(K/32)+c)*32 + lane)*16 + v]  holds  W[(c*32 + (lane>>4)*16 + v)*64 + w*16 + (lane&15)]
// so each lane's B fragment is one aligned 32-byte load.
template <int K>
__global__ void build_Bswz(const float* __restrict__ W, __bf16* __restrict__ Bs) {
    int idx = blockIdx.x * blockDim.x + threadIdx.x;
    if (idx >= K * HID) return;
    int v    = idx & 15;
    int l    = (idx >> 4) & 31;
    int rest = idx >> 9;
    int c    = rest % (K / 32);
    int w    = rest / (K / 32);
    int col  = (w << 4) | (l & 15);
    int kk   = c * 32 + ((l >> 4) << 4) + v;
    Bs[idx]  = (__bf16)W[kk * HID + col];
}

// ---------------------------------------------------------------- degree / norm
__global__ void deg_count(const int* __restrict__ dst, float* __restrict__ deg, int e) {
    int i = blockIdx.x * blockDim.x + threadIdx.x;
    if (i < e) atomicAdd(&deg[dst[i]], 1.0f);
}
__global__ void deg_to_dinv(float* __restrict__ deg, int n) {
    int i = blockIdx.x * blockDim.x + threadIdx.x;
    if (i < n) {
        float d = deg[i];                    // >= 1 (self loop)
        deg[i] = d > 0.0f ? rsqrtf(d) : 0.0f;
    }
}

// ---------------------------------------------------------------- WMMA GEMM
// C[N x 64] = A[N x K] (fp32 -> bf16 on the fly) @ W (pre-swizzled bf16 frags).
// 256 threads = 8 waves; each wave owns one 16-row tile and ALL four 16-col
// strips: A fragment loaded once per k-chunk, 4 back-to-back WMMAs.
template <int K>
__global__ __launch_bounds__(256) void gemm_wmma(const float* __restrict__ A,
                                                 const __bf16* __restrict__ Bs,
                                                 float* __restrict__ C) {
    const int lane = threadIdx.x & 31;
    const int wave = threadIdx.x >> 5;
    const int rowT = blockIdx.x * 8 + wave;           // wave-uniform
    if (rowT >= ROWT) return;                          // uniform exit: EXEC all-1 at WMMA
    const int m      = lane & 15;
    const int hiHalf = lane >> 4;
    const int kA     = hiHalf * 16;
    const float* Arow = A + (long)(rowT * 16 + m) * K;

    constexpr int CHUNKS      = K / 32;
    constexpr int STRIP_ELEMS = CHUNKS * 512;          // bf16 elems per col strip

    v8f acc[4] = {{}, {}, {}, {}};
#pragma unroll
    for (int c = 0; c < CHUNKS; ++c) {
        const int k0 = c * 32;
        if (c + 1 < CHUNKS) __builtin_prefetch(Arow + k0 + 32, 0, 0);
        // A fragment (ISA 7.12.2, 16-bit A 16x32)
        v16bf a;
#pragma unroll
        for (int j = 0; j < 4; ++j) {
            float2 lo = *(const float2*)(Arow + k0 + kA + 2 * j);
            float2 hi = *(const float2*)(Arow + k0 + kA + 8 + 2 * j);
            a[2 * j]         = (__bf16)lo.x;  a[2 * j + 1]     = (__bf16)lo.y;
            a[8 + 2 * j]     = (__bf16)hi.x;  a[8 + 2 * j + 1] = (__bf16)hi.y;
        }
        const __bf16* bp = Bs + (size_t)c * 512 + (size_t)lane * 16;
#pragma unroll
        for (int s = 0; s < 4; ++s) {
            v16bf b = *(const v16bf*)(bp + (size_t)s * STRIP_ELEMS);
            acc[s] = __builtin_amdgcn_wmma_f32_16x16x32_bf16(
                false, a, false, b, (short)0, acc[s], false, false);
        }
    }
    // C/D layout: VGPR r -> M = r + 8*hiHalf, N = lane&15 within each strip
    const int mBase = rowT * 16 + hiHalf * 8;
#pragma unroll
    for (int s = 0; s < 4; ++s) {
        const int col = (s << 4) | (lane & 15);
#pragma unroll
        for (int r = 0; r < 8; ++r)
            C[(long)(mBase + r) * HID + col] = acc[s][r];
    }
}

// ---------------------------------------------------------------- propagate
// self-loop (norm = dinv^2) + post-aggregation bias, float4-vectorized
__global__ void agg_init4(const float4* __restrict__ h, const float* __restrict__ dinv,
                          const float4* __restrict__ bias4, float4* __restrict__ out, int n4) {
    int idx = blockIdx.x * blockDim.x + threadIdx.x;
    if (idx >= n4) return;
    int i = idx >> 4;                                  // 16 float4 per node
    float s = dinv[i]; s *= s;
    float4 hv = h[idx];
    float4 bv = bias4[idx & 15];
    out[idx] = make_float4(hv.x * s + bv.x, hv.y * s + bv.y,
                           hv.z * s + bv.z, hv.w * s + bv.w);
}
// one thread per (edge, 4 features): 1 b128 gather + 4 L2-resident atomics
__global__ void agg_edges4(const int* __restrict__ ei, const float* __restrict__ h,
                           const float* __restrict__ dinv, float* __restrict__ out, int e) {
    int idx = blockIdx.x * blockDim.x + threadIdx.x;
    int ed = idx >> 4, q = idx & 15;
    if (ed >= e) return;
    int s = ei[ed], d = ei[EDGES + ed];
    float nrm = dinv[s] * dinv[d];
    float4 hv = *(const float4*)(h + s * HID + q * 4);
    float* o = out + d * HID + q * 4;
    atomicAdd(o + 0, hv.x * nrm);
    atomicAdd(o + 1, hv.y * nrm);
    atomicAdd(o + 2, hv.z * nrm);
    atomicAdd(o + 3, hv.w * nrm);
}

// ---------------------------------------------------------------- batch norm
// LDS block reduction, then one atomic per feature per block
__global__ void bn_stats(const float* __restrict__ h, float* __restrict__ sum,
                         float* __restrict__ sq, int n) {
    __shared__ float ls[256], lq[256];
    int f = threadIdx.x & 63, grp = threadIdx.x >> 6;
    int row = blockIdx.x * 4 + grp;
    int stride = gridDim.x * 4;
    float s = 0.0f, q = 0.0f;
    for (int i = row; i < n; i += stride) {
        float v = h[i * HID + f];
        s += v; q += v * v;
    }
    ls[threadIdx.x] = s; lq[threadIdx.x] = q;
    __syncthreads();
    if (grp == 0) {
        s = ls[f] + ls[64 + f] + ls[128 + f] + ls[192 + f];
        q = lq[f] + lq[64 + f] + lq[128 + f] + lq[192 + f];
        atomicAdd(&sum[f], s);
        atomicAdd(&sq[f], q);
    }
}
__global__ void bn_apply_relu4(const float4* __restrict__ h, const float* __restrict__ sum,
                               const float* __restrict__ sq, const float* __restrict__ gamma,
                               const float* __restrict__ beta, float4* __restrict__ out,
                               int n4, float invN) {
    int idx = blockIdx.x * blockDim.x + threadIdx.x;
    if (idx >= n4) return;
    int f4 = (idx & 15) * 4;
    float4 sm = *(const float4*)(sum + f4);
    float4 qq = *(const float4*)(sq + f4);
    float4 gm = *(const float4*)(gamma + f4);
    float4 bt = *(const float4*)(beta + f4);
    float4 hv = h[idx];
    float4 r;
#define BN1(cc) { float mean = sm.cc * invN;                              \
                  float var  = qq.cc * invN - mean * mean;                \
                  float aa   = gm.cc * rsqrtf(var + BN_EPS);              \
                  float y    = aa * (hv.cc - mean) + bt.cc;               \
                  r.cc = y > 0.0f ? y : 0.0f; }
    BN1(x) BN1(y) BN1(z) BN1(w)
#undef BN1
    out[idx] = r;
}

// ---------------------------------------------------------------- pooling + head
__global__ void pool4(const float4* __restrict__ h, const int* __restrict__ batch,
                      float* __restrict__ gsum, int* __restrict__ gmax,
                      int* __restrict__ gcnt, int n4) {
    int idx = blockIdx.x * blockDim.x + threadIdx.x;
    if (idx >= n4) return;
    int i = idx >> 4, q = idx & 15;
    int b = batch[i];
    float4 v = h[idx];                                 // >= 0 after ReLU
    float* gs = gsum + b * HID + q * 4;
    int*   gm = gmax + b * HID + q * 4;
    atomicAdd(gs + 0, v.x); atomicMax(gm + 0, __float_as_int(v.x));
    atomicAdd(gs + 1, v.y); atomicMax(gm + 1, __float_as_int(v.y));
    atomicAdd(gs + 2, v.z); atomicMax(gm + 2, __float_as_int(v.z));
    atomicAdd(gs + 3, v.w); atomicMax(gm + 3, __float_as_int(v.w));
    if (q == 0) atomicAdd(&gcnt[b], 1);
}
__global__ void head_kernel(const float* __restrict__ gsum, const int* __restrict__ gmax,
                            const int* __restrict__ gcnt, const float* __restrict__ Wout,
                            const float* __restrict__ bout, float* __restrict__ out, int g) {
    int i = blockIdx.x * blockDim.x + threadIdx.x;
    if (i >= g) return;
    float cnt = (float)gcnt[i];
    float inv = 1.0f / fmaxf(cnt, 1.0f);
    float acc = 0.0f;
#pragma unroll 4
    for (int f = 0; f < HID; ++f) {
        float mean = gsum[i * HID + f] * inv;
        float mx   = cnt > 0.0f ? __int_as_float(gmax[i * HID + f]) : 0.0f;
        acc += mean * Wout[f] + mx * Wout[HID + f];
    }
    out[i] = acc + bout[0];
}

// ---------------------------------------------------------------- launch
extern "C" void kernel_launch(void* const* d_in, const int* in_sizes, int n_in,
                              void* d_out, int out_size, void* d_ws, size_t ws_size,
                              hipStream_t stream) {
    const float* x    = (const float*)d_in[0];
    const int*   ei   = (const int*)d_in[1];       // [2, E] flat: src row 0, dst row 1
    const int*   bat  = (const int*)d_in[2];
    const float* W[3] = {(const float*)d_in[3], (const float*)d_in[5], (const float*)d_in[7]};
    const float* b[3] = {(const float*)d_in[4], (const float*)d_in[6], (const float*)d_in[8]};
    const float* g[3] = {(const float*)d_in[9],  (const float*)d_in[11], (const float*)d_in[13]};
    const float* be[3]= {(const float*)d_in[10], (const float*)d_in[12], (const float*)d_in[14]};
    const float* Wout = (const float*)d_in[15];
    const float* bout = (const float*)d_in[16];
    float* out = (float*)d_out;

    // workspace carve-up (64B aligned)
    char* ws = (char*)d_ws;
    size_t off = 0;
    auto carve = [&](size_t bytes) { char* p = ws + off; off += (bytes + 63) & ~size_t(63); return p; };
    float*  dinv  = (float*)carve(NODES * 4);
    float*  hA    = (float*)carve((size_t)NODES * HID * 4);
    float*  hB    = (float*)carve((size_t)NODES * HID * 4);
    __bf16* Bswz  = (__bf16*)carve((size_t)FIN * HID * 2);
    float*  bnsum = (float*)carve(HID * 4);
    float*  bnsq  = (float*)carve(HID * 4);
    float*  gsum  = (float*)carve((size_t)NGRAPH * HID * 4);
    int*    gmax  = (int*)carve((size_t)NGRAPH * HID * 4);
    int*    gcnt  = (int*)carve((size_t)NGRAPH * 4);

    const int NH  = NODES * HID;       // 6,400,000
    const int NH4 = NODES * 16;        // float4 count
    const int GH  = NGRAPH * HID;
    dim3 blk(256);
    auto gridFor = [](int n) { return dim3((unsigned)((n + 255) / 256)); };

    // degrees -> dinv (shared by all layers; self-loop contributes the 1.0 init)
    fill_f32<<<gridFor(NODES), blk, 0, stream>>>(dinv, 1.0f, NODES);
    deg_count<<<gridFor(EDGES), blk, 0, stream>>>(ei + EDGES, dinv, EDGES);
    deg_to_dinv<<<gridFor(NODES), blk, 0, stream>>>(dinv, NODES);

    const float invN = 1.0f / (float)NODES;
    const float* in = x;
    float* bufs[2] = {hA, hB};
    int flip = 0;
    for (int L = 0; L < 3; ++L) {
        float* gemmOut = bufs[flip];
        float* aggOut  = bufs[flip ^ 1];
        float* bnOut   = bufs[flip];
        int K = (L == 0) ? FIN : HID;

        if (K == FIN) {
            build_Bswz<FIN><<<gridFor(FIN * HID), blk, 0, stream>>>(W[L], Bswz);
            gemm_wmma<FIN><<<dim3((ROWT + 7) / 8), blk, 0, stream>>>(in, Bswz, gemmOut);
        } else {
            build_Bswz<HID><<<gridFor(HID * HID), blk, 0, stream>>>(W[L], Bswz);
            gemm_wmma<HID><<<dim3((ROWT + 7) / 8), blk, 0, stream>>>(in, Bswz, gemmOut);
        }

        agg_init4<<<gridFor(NH4), blk, 0, stream>>>((const float4*)gemmOut, dinv,
                                                    (const float4*)b[L], (float4*)aggOut, NH4);
        agg_edges4<<<dim3((unsigned)((size_t)EDGES * 16 / 256)), blk, 0, stream>>>(
            ei, gemmOut, dinv, aggOut, EDGES);

        fill_f32<<<dim3(1), dim3(128), 0, stream>>>(bnsum, 0.0f, HID);
        fill_f32<<<dim3(1), dim3(128), 0, stream>>>(bnsq, 0.0f, HID);
        bn_stats<<<dim3(512), blk, 0, stream>>>(aggOut, bnsum, bnsq, NODES);
        bn_apply_relu4<<<gridFor(NH4), blk, 0, stream>>>((const float4*)aggOut, bnsum, bnsq,
                                                         g[L], be[L], (float4*)bnOut, NH4, invN);
        in = bnOut;
        flip ^= 1;
    }
    const float* hFinal = in;

    // pooling + head
    fill_f32<<<gridFor(GH), blk, 0, stream>>>(gsum, 0.0f, GH);
    fill_i32<<<gridFor(GH), blk, 0, stream>>>(gmax, 0, GH);
    fill_i32<<<gridFor(NGRAPH), blk, 0, stream>>>(gcnt, 0, NGRAPH);
    pool4<<<gridFor(NH4), blk, 0, stream>>>((const float4*)hFinal, bat, gsum, gmax, gcnt, NH4);
    head_kernel<<<gridFor(NGRAPH), dim3(128), 0, stream>>>(gsum, gmax, gcnt, Wout, bout,
                                                           out, NGRAPH);
}
